// RNNCell_47802986004975
// MI455X (gfx1250) — matrix operand order repaired and checked by reading, not compiled, not run on hardware
//
#include <hip/hip_runtime.h>
#include <hip/hip_bf16.h>

// CDNA5 / gfx1250 tanh-RNN, B=64 T=512 I=H=1024.
//  Phase 0: convert W_x, W_h to bf16; zero h0 and the device-barrier counter.
//  Phase 1: xW[t][b][:] = x[b][t][:] @ W_x^T + bias via v_wmma_f32_16x16x32_bf16
//           (bandwidth-bound; x converted bf16 in-register, re-read by each of 16 column blocks).
//  Phase 2: ONE persistent kernel (32 blocks x 256 thr), all 512 timesteps:
//           - W_h slice (32 cols x 1024 K, 64KB bf16) preloaded to LDS once,
//             padded stride 1032 to spread banks; B-fragments via ds_load_b128.
//           - h double-buffered in global; device-wide barrier per step via
//             monotonic atomic counter (release add / acquire spin + s_sleep).

typedef __attribute__((ext_vector_type(16))) __bf16 v16bf;
typedef __attribute__((ext_vector_type(8)))  float  v8f;
typedef __attribute__((ext_vector_type(4)))  __bf16 v4bf;

#define BB 64
#define TT 512
#define II 1024
#define HH 1024
#define LDS_STRIDE 1032   // 1024 + 8 bf16 pad -> 2064B rows, distinct bank groups

union Frag16 { v16bf v; uint4 q[2]; };

// A-matrix 16x32 bf16 fragment (ISA 7.12.2): lanes 0-15 row M=lane, K=k0+{0..7,16..23};
// lanes 16-31 row M=lane-16, K=k0+{8..15,24..31}.
__device__ inline v16bf load_afrag_bf16(const __bf16* __restrict__ base, int lda,
                                        int m0, int k0, int lane) {
    const int r    = lane & 15;
    const int koff = (lane >> 4) * 8;
    const __bf16* p = base + (size_t)(m0 + r) * lda + k0 + koff;
    Frag16 f;
    f.q[0] = *(const uint4*)(p);
    f.q[1] = *(const uint4*)(p + 16);
    return f.v;
}

// B-matrix 32x16 fragment from global: B = W^T, column N of B = row N of W.
__device__ inline v16bf load_bfrag_bf16(const __bf16* __restrict__ W, int ldw,
                                        int n0, int k0, int lane) {
    const int c    = lane & 15;
    const int koff = (lane >> 4) * 8;
    const __bf16* p = W + (size_t)(n0 + c) * ldw + k0 + koff;
    Frag16 f;
    f.q[0] = *(const uint4*)(p);
    f.q[1] = *(const uint4*)(p + 16);
    return f.v;
}

// Same B-fragment but from LDS (padded stride) -> ds_load_b128 x2.
__device__ inline v16bf load_bfrag_lds(const __bf16* lds, int nLocal, int k0, int lane) {
    const int c    = lane & 15;
    const int koff = (lane >> 4) * 8;
    const __bf16* p = lds + (nLocal + c) * LDS_STRIDE + k0 + koff;
    Frag16 f;
    f.q[0] = *(const uint4*)(p);
    f.q[1] = *(const uint4*)(p + 16);
    return f.v;
}

__global__ __launch_bounds__(256)
void cvt_f32_to_bf16(const float* __restrict__ src, __bf16* __restrict__ dst, int n4) {
    int i = blockIdx.x * blockDim.x + threadIdx.x;
    if (i < n4) {
        float4 v = ((const float4*)src)[i];
        v4bf o;
        o[0] = (__bf16)v.x; o[1] = (__bf16)v.y; o[2] = (__bf16)v.z; o[3] = (__bf16)v.w;
        ((v4bf*)dst)[i] = o;
    }
}

__global__ __launch_bounds__(256)
void zero_bytes16(uint4* __restrict__ p) {
    p[blockIdx.x * blockDim.x + threadIdx.x] = make_uint4(0u, 0u, 0u, 0u);
}

// ---------------- Phase 1: xW = x @ Wx^T + bias, stored [T][B][H] ----------------
__global__ __launch_bounds__(128)
void gemm_xw(const float* __restrict__ x, const __bf16* __restrict__ Wx,
             const float* __restrict__ bias, float* __restrict__ xW) {
    const int lane = threadIdx.x & 31;
    const int wave = threadIdx.x >> 5;
    const int n0 = blockIdx.x * 64;
    const int m0 = blockIdx.y * 64 + wave * 16;

    v8f acc[4] = {v8f{}, v8f{}, v8f{}, v8f{}};

    const int koffA = (lane >> 4) * 8;
    const float* ap = x + (size_t)(m0 + (lane & 15)) * II + koffA;

    for (int k0 = 0; k0 < II; k0 += 32) {
        float4 a0 = *(const float4*)(ap + k0);
        float4 a1 = *(const float4*)(ap + k0 + 4);
        float4 a2 = *(const float4*)(ap + k0 + 16);
        float4 a3 = *(const float4*)(ap + k0 + 20);
        v16bf af;
        af[0]  = (__bf16)a0.x; af[1]  = (__bf16)a0.y; af[2]  = (__bf16)a0.z; af[3]  = (__bf16)a0.w;
        af[4]  = (__bf16)a1.x; af[5]  = (__bf16)a1.y; af[6]  = (__bf16)a1.z; af[7]  = (__bf16)a1.w;
        af[8]  = (__bf16)a2.x; af[9]  = (__bf16)a2.y; af[10] = (__bf16)a2.z; af[11] = (__bf16)a2.w;
        af[12] = (__bf16)a3.x; af[13] = (__bf16)a3.y; af[14] = (__bf16)a3.z; af[15] = (__bf16)a3.w;

        v16bf b0 = load_bfrag_bf16(Wx, II, n0 +  0, k0, lane);
        v16bf b1 = load_bfrag_bf16(Wx, II, n0 + 16, k0, lane);
        v16bf b2 = load_bfrag_bf16(Wx, II, n0 + 32, k0, lane);
        v16bf b3 = load_bfrag_bf16(Wx, II, n0 + 48, k0, lane);

        acc[0] = __builtin_amdgcn_wmma_f32_16x16x32_bf16(false, af, false, b0, (short)0, acc[0], false, false);
        acc[1] = __builtin_amdgcn_wmma_f32_16x16x32_bf16(false, af, false, b1, (short)0, acc[1], false, false);
        acc[2] = __builtin_amdgcn_wmma_f32_16x16x32_bf16(false, af, false, b2, (short)0, acc[2], false, false);
        acc[3] = __builtin_amdgcn_wmma_f32_16x16x32_bf16(false, af, false, b3, (short)0, acc[3], false, false);
    }

    const int col   = lane & 15;
    const int rbase = m0 + (lane >> 4) * 8;
#pragma unroll
    for (int nt = 0; nt < 4; ++nt) {
        const int n = n0 + nt * 16 + col;
        const float bv = bias[n];
        v8f a = acc[nt];
#pragma unroll
        for (int j = 0; j < 8; ++j) {
            const int mm   = rbase + j;
            const int bidx = mm >> 9;    // / T
            const int tidx = mm & 511;   // % T
            xW[((size_t)tidx * BB + bidx) * HH + n] = a[j] + bv;
        }
    }
}

// ---------------- Phase 2: persistent recurrence kernel ----------------
// 32 blocks x 256 threads (8 waves). Block b owns output columns [b*32, b*32+32).
// Wave w: m0 = (w&3)*16 (batch rows), nLocal = (w>>2)*16 (column sub-tile).
__global__ __launch_bounds__(256)
void rnn_persistent(const __bf16* __restrict__ Wh, const float* __restrict__ xW,
                    float* __restrict__ out, __bf16* __restrict__ h0,
                    __bf16* __restrict__ h1, unsigned* __restrict__ barrier_ctr) {
    __shared__ __bf16 lWh[32 * LDS_STRIDE];   // ~66 KB of the 320 KB WGP LDS

    const int lane = threadIdx.x & 31;
    const int wave = threadIdx.x >> 5;
    const int nbase  = blockIdx.x * 32;       // global column base of this block
    const int m0     = (wave & 3) * 16;       // batch-row tile
    const int nLocal = (wave >> 2) * 16;      // 0 or 16 within block's 32 cols

    // One-time preload: Wh rows [nbase, nbase+32) x K=1024 into padded LDS.
    for (int c = threadIdx.x; c < 32 * 128; c += 256) {
        const int row = c >> 7, chunk = c & 127;
        uint4 v = *(const uint4*)(Wh + (size_t)(nbase + row) * HH + chunk * 8);
        *(uint4*)(&lWh[row * LDS_STRIDE + chunk * 8]) = v;
    }
    __syncthreads();

    const int col   = nbase + nLocal + (lane & 15);
    const int rbase = m0 + (lane >> 4) * 8;
    __bf16* hb[2] = {h0, h1};

    for (int t = 0; t < TT; ++t) {
        const __bf16* hprev = hb[t & 1];
        __bf16*       hnext = hb[(t + 1) & 1];

        v8f acc = v8f{};
        for (int k0 = 0; k0 < HH; k0 += 32) {
            v16bf af = load_afrag_bf16(hprev, HH, m0, k0, lane);
            v16bf bf = load_bfrag_lds(lWh, nLocal, k0, lane);
            acc = __builtin_amdgcn_wmma_f32_16x16x32_bf16(false, af, false, bf, (short)0, acc, false, false);
        }

        const float* xwp = xW + (size_t)t * (BB * HH);
#pragma unroll
        for (int j = 0; j < 8; ++j) {
            const int b = rbase + j;
            const float y = tanhf(xwp[(size_t)b * HH + col] + acc[j]);
            out[((size_t)b * TT + t) * HH + col] = y;
            hnext[(size_t)b * HH + col] = (__bf16)y;
        }

        // Device-wide barrier: monotonic counter, release on arrive, acquire spin.
        __syncthreads();
        if (threadIdx.x == 0) {
            __hip_atomic_fetch_add(barrier_ctr, 1u, __ATOMIC_RELEASE, __HIP_MEMORY_SCOPE_AGENT);
            const unsigned target = 32u * (unsigned)(t + 1);
            while (__hip_atomic_load(barrier_ctr, __ATOMIC_ACQUIRE, __HIP_MEMORY_SCOPE_AGENT) < target) {
                __builtin_amdgcn_s_sleep(1);
            }
        }
        __syncthreads();
        __threadfence();   // acquire: make other WGPs' hnext stores visible (inv WGP$)
    }
}

extern "C" void kernel_launch(void* const* d_in, const int* in_sizes, int n_in,
                              void* d_out, int out_size, void* d_ws, size_t ws_size,
                              hipStream_t stream) {
    const float* x    = (const float*)d_in[0];   // [B,T,I]
    const float* Wx   = (const float*)d_in[1];   // [H,I]
    const float* Wh   = (const float*)d_in[2];   // [H,H]
    const float* bias = (const float*)d_in[3];   // [H]
    float* out = (float*)d_out;                  // [B,T,H]

    char* ws = (char*)d_ws;
    __bf16*   wsWx = (__bf16*)(ws);                                 // 2 MB
    __bf16*   wsWh = (__bf16*)(ws + (2u << 20));                    // 2 MB
    __bf16*   h0   = (__bf16*)(ws + (4u << 20));                    // 128 KB
    __bf16*   h1   = (__bf16*)(ws + (4u << 20) + (256u << 10));     // 128 KB
    unsigned* ctr  = (unsigned*)(ws + (4u << 20) + (512u << 10));   // 16 B
    float*    xw   = (float*) (ws + (8u << 20));                    // 128 MB [T][B][H]

    // Phase 0
    cvt_f32_to_bf16<<<(II * HH / 4 + 255) / 256, 256, 0, stream>>>(Wx, wsWx, II * HH / 4);
    cvt_f32_to_bf16<<<(HH * HH / 4 + 255) / 256, 256, 0, stream>>>(Wh, wsWh, HH * HH / 4);
    zero_bytes16<<<(BB * HH * 2 / 16) / 256, 256, 0, stream>>>((uint4*)h0);
    zero_bytes16<<<1, 1, 0, stream>>>((uint4*)ctr);

    // Phase 1
    gemm_xw<<<dim3(HH / 64, (BB * TT) / 64), 128, 0, stream>>>(x, wsWx, bias, xw);

    // Phase 2: one persistent launch for all 512 steps.
    rnn_persistent<<<32, 256, 0, stream>>>(wsWh, xw, out, h0, h1, ctr);
}
